// Transformer_16879221473953
// MI455X (gfx1250) — compile-verified, matrix operand order checked by reading, not run on hardware
//
#include <hip/hip_runtime.h>
#include <cstdint>

// ---------------------------------------------------------------------------
// Transformer forward for MI455X (gfx1250, wave32, WMMA + async-LDS DMA).
//  - All GEMMs: v_wmma_f32_16x16x32_bf16, f32 accumulate.
//  - Weights pre-packed once per launch into bf16 [n][k] 128x32 LDS-shaped
//    tiles so the GEMM B feed is a pure global_load_async_to_lds_b128 DMA.
//  - A tiles async-copied as raw f32, converted at fragment build
//    (v_cvt_pk_bf16_f32), double-buffered with s_wait_asynccnt.
// ---------------------------------------------------------------------------

#define E_DIM   512
#define H_HEADS 8
#define HD_DIM  64
#define L_LAYERS 6
#define FF_DIM  2048
#define FC_DIM  2048
#define V_SIZE  10000
#define S_LEN   196
#define NB      64
#define TT      48

typedef __bf16 v16bf __attribute__((ext_vector_type(16)));
typedef float  v8f   __attribute__((ext_vector_type(8)));

union Frag {
  v16bf v;
  unsigned int u[8];
  unsigned short h[16];
};

__device__ __forceinline__ unsigned short bf16_bits(float f) {
  __bf16 b = (__bf16)f;
  return __builtin_bit_cast(unsigned short, b);
}
__device__ __forceinline__ unsigned int pack2_bf16(float x, float y) {
  return (unsigned int)bf16_bits(x) | ((unsigned int)bf16_bits(y) << 16);
}

// ISA 7.12.2: 16-bit A-matrix 16x32 — lane[3:0]=M row, lane[4] = K half (+8).
__device__ __forceinline__ int koff_elem(int khalf, int j) {
  return (j < 8) ? (khalf * 8 + j) : (16 + khalf * 8 + (j - 8));
}
__device__ __forceinline__ int koff_pair(int khalf, int i) {  // K of uint pair i
  return (i < 4) ? (khalf * 8 + 2 * i) : (16 + khalf * 8 + 2 * (i - 4));
}

// gfx1250 async global->LDS copy, 16B per lane, tracked by ASYNCcnt.
// Generic pointers: low 32 bits of a flat LDS address are the LDS offset
// (ISA 10.2 aperture rules), so truncation yields the DS address VGPR.
__device__ __forceinline__ void async_b128(const void* g, void* l) {
  unsigned lds = (unsigned)(uintptr_t)l;
  asm volatile("global_load_async_to_lds_b128 %0, %1, off"
               :: "v"(lds), "v"(g) : "memory");
}
#define WAIT_ASYNC0 asm volatile("s_wait_asynccnt 0x0" ::: "memory")
#define WAIT_ASYNC6 asm volatile("s_wait_asynccnt 0x6" ::: "memory")

// ---------------------------------------------------------------------------
// Weight pre-pack: f32 [K][N] -> bf16 tiles [ntile][K/32][n:128][k:32].
// One 8KB tile per block; padding columns zero-filled.
// ---------------------------------------------------------------------------
__global__ __launch_bounds__(256) void pack_weight_kernel(
    const float* __restrict__ W, unsigned short* __restrict__ P, int K, int N)
{
  const int kt = blockIdx.x;              // k0/32
  const int nt = blockIdx.y;              // n tile
  const int ktiles = K >> 5;
  const size_t base = ((size_t)nt * ktiles + kt) * 4096;
  for (int i = 0; i < 16; ++i) {
    int e = threadIdx.x + 256 * i;        // 4096 elements: [n][k]
    int n = e >> 5, k = e & 31;
    int gn = nt * 128 + n;
    float v = (gn < N) ? W[(size_t)(kt * 32 + k) * N + gn] : 0.f;
    P[base + e] = bf16_bits(v);
  }
}

// ---------------------------------------------------------------------------
// GEMM: C[M,N] = act(A[M,K] @ B + bias). A f32 HBM, B pre-packed bf16 tiles.
// Block tile 128x128, K-step 32, double-buffered async-LDS staging.
// 8 waves: 4 along M (32 rows) x 2 along N (64 cols), 8 WMMA accs each.
// ---------------------------------------------------------------------------
__global__ __launch_bounds__(256) void wmma_gemm_kernel(
    const float* __restrict__ A, const unsigned short* __restrict__ Bp,
    const float* __restrict__ bias, float* __restrict__ C,
    int M, int N, int K, int relu)
{
  __shared__ float  Asf[2][128 * 32];     // raw f32 A tiles
  __shared__ __bf16 Bs[2][128 * 32];      // packed bf16 B tiles [n][k]

  const int tid    = threadIdx.x;
  const int lane   = tid & 31;
  const int wid    = tid >> 5;
  const int wm     = wid & 3;
  const int wn     = wid >> 2;
  const int m_tile = blockIdx.y * 128;
  const int khalf  = lane >> 4;
  const int l16    = lane & 15;
  const int ktiles = K >> 5;
  const size_t btile0 = (size_t)blockIdx.x * ktiles * 4096;

  const v8f vzero = {0.f, 0.f, 0.f, 0.f, 0.f, 0.f, 0.f, 0.f};
  v8f acc[2][4];
  for (int ms = 0; ms < 2; ++ms)
    for (int ns = 0; ns < 4; ++ns)
      acc[ms][ns] = vzero;

  auto issue = [&](int t, int buf) {
    // A tile: 128x32 f32 = 16KB = 1024 x 16B chunks (4 per thread).
    for (int i = 0; i < 4; ++i) {
      int c = tid + 256 * i;
      int m = c >> 3, c4 = c & 7;
      int gm = m_tile + m;
      if (gm >= M) gm = M - 1;            // clamp: keep EXEC all-ones
      async_b128(&A[(size_t)gm * K + t * 32 + c4 * 4], &Asf[buf][m * 32 + c4 * 4]);
    }
    // B tile: 8KB contiguous = 512 x 16B chunks (2 per thread).
    const unsigned short* bt = Bp + btile0 + (size_t)t * 4096;
    for (int i = 0; i < 2; ++i) {
      int c = tid + 256 * i;
      async_b128(&bt[c * 8], &Bs[buf][c * 8]);
    }
  };

  issue(0, 0);
  for (int t = 0; t < ktiles; ++t) {
    const int buf = t & 1;
    if (t + 1 < ktiles) { issue(t + 1, buf ^ 1); WAIT_ASYNC6; }
    else                { WAIT_ASYNC0; }
    __syncthreads();

    Frag bf[4];
    for (int ns = 0; ns < 4; ++ns) {
      int n = wn * 64 + ns * 16 + l16;
      const unsigned int* row = reinterpret_cast<const unsigned int*>(&Bs[buf][n * 32]);
      for (int i = 0; i < 8; ++i)
        bf[ns].u[i] = row[koff_pair(khalf, i) >> 1];
    }
    for (int ms = 0; ms < 2; ++ms) {
      Frag af;
      int m = wm * 32 + ms * 16 + l16;
      const float* arow = &Asf[buf][m * 32];
      for (int i = 0; i < 8; ++i) {
        float2 f = *reinterpret_cast<const float2*>(&arow[koff_pair(khalf, i)]);
        af.u[i] = pack2_bf16(f.x, f.y);
      }
      for (int ns = 0; ns < 4; ++ns)
        acc[ms][ns] = __builtin_amdgcn_wmma_f32_16x16x32_bf16(
            false, af.v, false, bf[ns].v, (short)0, acc[ms][ns], false, false);
    }
    __syncthreads();
  }

  const int rbase = khalf * 8;
  for (int ms = 0; ms < 2; ++ms)
    for (int ns = 0; ns < 4; ++ns) {
      int gn = blockIdx.x * 128 + wn * 64 + ns * 16 + l16;
      if (gn >= N) continue;
      float bv = bias ? bias[gn] : 0.f;
      for (int i = 0; i < 8; ++i) {
        int gm = m_tile + wm * 32 + ms * 16 + rbase + i;
        if (gm >= M) continue;
        float r = acc[ms][ns][i] + bv;
        if (relu) r = fmaxf(r, 0.f);
        C[(size_t)gm * N + gn] = r;
      }
    }
}

// ---------------------------------------------------------------------------
// Fused attention, one wave per (batch, head, 16-query block).
// Q/K/V tiles async-staged to LDS as f32; scores+softmax in LDS; both
// GEMMs (Q.Kt and P.V) via WMMA bf16.
// ---------------------------------------------------------------------------
__global__ __launch_bounds__(32) void attn_kernel(
    const float* __restrict__ Q, const float* __restrict__ K,
    const float* __restrict__ V, float* __restrict__ O,
    int q_len, int k_len, int causal)
{
  __shared__ float  sc[16][224];          // raw energies
  __shared__ __bf16 pb[16][224];          // probabilities (zero padded)
  __shared__ float  Kt[16 * 64];          // Q tile, then K tiles
  __shared__ float  Vt[32 * 64];          // V tiles

  const int lane  = threadIdx.x;
  const int l16   = lane & 15;
  const int khalf = lane >> 4;
  const int qb    = blockIdx.x * 16;
  const int h     = blockIdx.y;
  const int n     = blockIdx.z;
  const int hofs  = h * HD_DIM;

  // ---- Q fragments (16x64): async stage + packed convert ----
  Frag qf[2];
  {
    for (int i = 0; i < 8; ++i) {
      int c = lane + 32 * i;              // 256 x 16B chunks
      int r = c >> 2, c4 = c & 3;
      int m = qb + r;
      if (m >= q_len) m = q_len - 1;
      async_b128(&Q[(size_t)(n * q_len + m) * E_DIM + hofs + c4 * 4],
                 &Kt[r * 64 + c4 * 4]);
    }
    WAIT_ASYNC0;
    for (int s = 0; s < 2; ++s)
      for (int i = 0; i < 8; ++i) {
        float2 f = *reinterpret_cast<const float2*>(
            &Kt[l16 * 64 + s * 32 + koff_pair(khalf, i)]);
        qf[s].u[i] = pack2_bf16(f.x, f.y);
      }
  }

  const v8f vzero = {0.f, 0.f, 0.f, 0.f, 0.f, 0.f, 0.f, 0.f};
  const int nkb = (k_len + 15) >> 4;
  for (int kb = 0; kb < nkb; ++kb) {
    for (int i = 0; i < 8; ++i) {
      int c = lane + 32 * i;
      int r = c >> 2, c4 = c & 3;
      int key = kb * 16 + r;
      if (key >= k_len) key = k_len - 1;
      async_b128(&K[(size_t)(n * k_len + key) * E_DIM + hofs + c4 * 4],
                 &Kt[r * 64 + c4 * 4]);
    }
    WAIT_ASYNC0;
    v8f e = vzero;
    for (int s = 0; s < 2; ++s) {
      Frag kf;
      for (int i = 0; i < 8; ++i) {
        float2 f = *reinterpret_cast<const float2*>(
            &Kt[l16 * 64 + s * 32 + koff_pair(khalf, i)]);
        kf.u[i] = pack2_bf16(f.x, f.y);
      }
      e = __builtin_amdgcn_wmma_f32_16x16x32_bf16(
          false, qf[s].v, false, kf.v, (short)0, e, false, false);
    }
    const int rbase = khalf * 8;
    const int kcol = kb * 16 + l16;
    for (int i = 0; i < 8; ++i) {
      int r = rbase + i;
      float ev = e[i];
      if (kcol >= k_len) ev = -3.0e38f;                 // padded key
      else if (causal && kcol > qb + r) ev = -1.0e20f;  // tril mask fill
      sc[r][kcol] = ev;
    }
  }
  __syncthreads();

  // ---- softmax over keys, reference scale 1/sqrt(E) ----
  if (lane < 16) {
    const float scale = 0.04419417382415922f;  // 1/sqrt(512)
    float mx = -3.0e38f;
    for (int k = 0; k < k_len; ++k) mx = fmaxf(mx, sc[lane][k]);
    float sum = 0.f;
    for (int k = 0; k < k_len; ++k) sum += __expf((sc[lane][k] - mx) * scale);
    float inv = 1.f / sum;
    for (int k = 0; k < 224; ++k) {
      float p = (k < k_len) ? __expf((sc[lane][k] - mx) * scale) * inv : 0.f;
      pb[lane][k] = (__bf16)p;
    }
  }
  __syncthreads();

  // ---- ctx = P (16 x k_len) @ V (k_len x 64) ----
  v8f o[4];
  for (int ns = 0; ns < 4; ++ns) o[ns] = vzero;
  const int ksteps = (k_len + 31) >> 5;
  for (int ks = 0; ks < ksteps; ++ks) {
    for (int i = 0; i < 16; ++i) {
      int c = lane + 32 * i;              // 512 x 16B chunks (32x64 f32)
      int r = c >> 2, c4 = c & 3;
      int kv = ks * 32 + r;
      if (kv >= k_len) kv = k_len - 1;    // P is 0 there, garbage*0 = 0
      async_b128(&V[(size_t)(n * k_len + kv) * E_DIM + hofs + c4 * 4],
                 &Vt[r * 64 + c4 * 4]);
    }
    WAIT_ASYNC0;
    Frag af;
    const unsigned short* prow = reinterpret_cast<const unsigned short*>(&pb[l16][0]);
    for (int i = 0; i < 8; ++i)
      af.u[i] = *reinterpret_cast<const unsigned int*>(
          &prow[ks * 32 + koff_pair(khalf, i)]);
    for (int ns = 0; ns < 4; ++ns) {
      Frag vf;
      for (int j = 0; j < 16; ++j) {
        int kvl = koff_elem(khalf, j);    // local 0..31
        vf.h[j] = bf16_bits(Vt[kvl * 64 + ns * 16 + l16]);
      }
      o[ns] = __builtin_amdgcn_wmma_f32_16x16x32_bf16(
          false, af.v, false, vf.v, (short)0, o[ns], false, false);
    }
  }

  const int rbase = khalf * 8;
  for (int ns = 0; ns < 4; ++ns)
    for (int i = 0; i < 8; ++i) {
      int m = qb + rbase + i;
      if (m < q_len)
        O[(size_t)(n * q_len + m) * E_DIM + hofs + ns * 16 + l16] = o[ns][i];
    }
}

// ---------------------------------------------------------------------------
// out = layernorm(a + b) * g + beta   (row length 512, one wave per row)
// ---------------------------------------------------------------------------
__global__ __launch_bounds__(256) void ln_add_kernel(
    const float* __restrict__ a, const float* __restrict__ b,
    const float* __restrict__ g, const float* __restrict__ beta,
    float* __restrict__ out, int rows)
{
  int lane = threadIdx.x & 31;
  int wid  = threadIdx.x >> 5;
  int row  = blockIdx.x * 8 + wid;
  if (row >= rows) return;
  size_t base = (size_t)row * E_DIM;
  float x[16];
  float s = 0.f;
  for (int i = 0; i < 16; ++i) {
    int c = lane + 32 * i;
    x[i] = a[base + c] + b[base + c];
    s += x[i];
  }
  for (int off = 16; off > 0; off >>= 1) s += __shfl_xor(s, off, 32);
  float mean = s * (1.f / E_DIM);
  float v = 0.f;
  for (int i = 0; i < 16; ++i) { float d = x[i] - mean; v += d * d; }
  for (int off = 16; off > 0; off >>= 1) v += __shfl_xor(v, off, 32);
  float inv = rsqrtf(v * (1.f / E_DIM) + 1e-5f);
  for (int i = 0; i < 16; ++i) {
    int c = lane + 32 * i;
    out[base + c] = (x[i] - mean) * inv * g[c] + beta[c];
  }
}

__global__ void add_pos_kernel(float* __restrict__ x, const float* __restrict__ pos,
                               int rows, int period)
{
  int idx = blockIdx.x * blockDim.x + threadIdx.x;
  if (idx >= rows * E_DIM) return;
  int r = idx / E_DIM, c = idx - r * E_DIM;
  x[idx] += pos[(r % period) * E_DIM + c];
}

__global__ void embed_kernel(const int* __restrict__ trg, const float* __restrict__ emb,
                             const float* __restrict__ dpos, float* __restrict__ y, int total)
{
  int idx = blockIdx.x * blockDim.x + threadIdx.x;
  if (idx >= total * E_DIM) return;
  int rt = idx / E_DIM, c = idx - rt * E_DIM;
  int t = rt % TT;
  int tok = trg[rt];
  y[idx] = emb[(size_t)tok * E_DIM + c] * 22.627416997969522f + dpos[t * E_DIM + c];
}

// ---------------------------------------------------------------------------
// Host orchestration
// ---------------------------------------------------------------------------
extern "C" void kernel_launch(void* const* d_in, const int* in_sizes, int n_in,
                              void* d_out, int out_size, void* d_ws, size_t ws_size,
                              hipStream_t stream)
{
  (void)in_sizes; (void)n_in; (void)out_size; (void)ws_size;

  const float* src      = (const float*)d_in[0];
  const int*   trg      = (const int*)  d_in[1];
  const float* fc_w     = (const float*)d_in[2];
  const float* fc_b     = (const float*)d_in[3];
  const float* enc_pos  = (const float*)d_in[4];
  const float* e_wq     = (const float*)d_in[5];
  const float* e_wk     = (const float*)d_in[6];
  const float* e_wv     = (const float*)d_in[7];
  const float* e_wo     = (const float*)d_in[8];
  const float* e_bo     = (const float*)d_in[9];
  const float* e_ln1g   = (const float*)d_in[10];
  const float* e_ln1b   = (const float*)d_in[11];
  const float* e_ln2g   = (const float*)d_in[12];
  const float* e_ln2b   = (const float*)d_in[13];
  const float* e_ffw1   = (const float*)d_in[14];
  const float* e_ffb1   = (const float*)d_in[15];
  const float* e_ffw2   = (const float*)d_in[16];
  const float* e_ffb2   = (const float*)d_in[17];
  const float* word_emb = (const float*)d_in[18];
  const float* dec_pos  = (const float*)d_in[19];
  const float* s_wq     = (const float*)d_in[20];
  const float* s_wk     = (const float*)d_in[21];
  const float* s_wv     = (const float*)d_in[22];
  const float* s_wo     = (const float*)d_in[23];
  const float* s_bo     = (const float*)d_in[24];
  const float* s_lng    = (const float*)d_in[25];
  const float* s_lnb    = (const float*)d_in[26];
  const float* c_wq     = (const float*)d_in[27];
  const float* c_wk     = (const float*)d_in[28];
  const float* c_wv     = (const float*)d_in[29];
  const float* c_wo     = (const float*)d_in[30];
  const float* c_bo     = (const float*)d_in[31];
  const float* c_ln1g   = (const float*)d_in[32];
  const float* c_ln1b   = (const float*)d_in[33];
  const float* c_ln2g   = (const float*)d_in[34];
  const float* c_ln2b   = (const float*)d_in[35];
  const float* c_ffw1   = (const float*)d_in[36];
  const float* c_ffb1   = (const float*)d_in[37];
  const float* c_ffw2   = (const float*)d_in[38];
  const float* c_ffb2   = (const float*)d_in[39];
  const float* out_w    = (const float*)d_in[40];
  const float* out_b    = (const float*)d_in[41];
  float* out = (float*)d_out;

  const int Me = NB * S_LEN;   // 12544
  const int Md = NB * TT;      // 3072
  const size_t EE = (size_t)E_DIM * E_DIM;
  const size_t EF = (size_t)E_DIM * FF_DIM;

  float* ws = (float*)d_ws;
  size_t off = 0;
  auto alloc = [&](size_t nf) { float* p = ws + off; off += nf; return p; };
  float* x0  = alloc((size_t)Me * E_DIM);
  float* bq  = alloc((size_t)Me * E_DIM);
  float* bk  = alloc((size_t)Me * E_DIM);
  float* bv  = alloc((size_t)Me * E_DIM);
  float* ba  = alloc((size_t)Me * E_DIM);
  float* bb  = alloc((size_t)Me * E_DIM);
  float* bx1 = alloc((size_t)Me * E_DIM);
  float* ffh = alloc((size_t)Me * FF_DIM);
  float* ffo = alloc((size_t)Me * E_DIM);
  float* dy  = alloc((size_t)Md * E_DIM);

  // bf16 packed-weight arena (16B aligned: all counts above are x512 floats)
  unsigned short* wsu = (unsigned short*)(ws + off);
  size_t uoff = 0;
  auto ualloc = [&](size_t n) { unsigned short* p = wsu + uoff; uoff += n; return p; };
  auto packed_elems = [](int K, int N) {
    return (size_t)K * (size_t)((N + 127) / 128) * 128;
  };
  auto pack = [&](const float* W, unsigned short* P, int K, int N) {
    dim3 grid(K / 32, (N + 127) / 128);
    pack_weight_kernel<<<grid, dim3(256), 0, stream>>>(W, P, K, N);
  };

  unsigned short *P_fcw, *P_outw;
  unsigned short *P_ewq[L_LAYERS], *P_ewk[L_LAYERS], *P_ewv[L_LAYERS], *P_ewo[L_LAYERS];
  unsigned short *P_ef1[L_LAYERS], *P_ef2[L_LAYERS];
  unsigned short *P_swq[L_LAYERS], *P_swk[L_LAYERS], *P_swv[L_LAYERS], *P_swo[L_LAYERS];
  unsigned short *P_cwq[L_LAYERS], *P_cwk[L_LAYERS], *P_cwv[L_LAYERS], *P_cwo[L_LAYERS];
  unsigned short *P_cf1[L_LAYERS], *P_cf2[L_LAYERS];

  P_fcw = ualloc(packed_elems(FC_DIM, E_DIM)); pack(fc_w, P_fcw, FC_DIM, E_DIM);
  for (int i = 0; i < L_LAYERS; ++i) {
    P_ewq[i] = ualloc(EE); pack(e_wq + i * EE, P_ewq[i], E_DIM, E_DIM);
    P_ewk[i] = ualloc(EE); pack(e_wk + i * EE, P_ewk[i], E_DIM, E_DIM);
    P_ewv[i] = ualloc(EE); pack(e_wv + i * EE, P_ewv[i], E_DIM, E_DIM);
    P_ewo[i] = ualloc(EE); pack(e_wo + i * EE, P_ewo[i], E_DIM, E_DIM);
    P_ef1[i] = ualloc(EF); pack(e_ffw1 + i * EF, P_ef1[i], E_DIM, FF_DIM);
    P_ef2[i] = ualloc(EF); pack(e_ffw2 + i * EF, P_ef2[i], FF_DIM, E_DIM);
    P_swq[i] = ualloc(EE); pack(s_wq + i * EE, P_swq[i], E_DIM, E_DIM);
    P_swk[i] = ualloc(EE); pack(s_wk + i * EE, P_swk[i], E_DIM, E_DIM);
    P_swv[i] = ualloc(EE); pack(s_wv + i * EE, P_swv[i], E_DIM, E_DIM);
    P_swo[i] = ualloc(EE); pack(s_wo + i * EE, P_swo[i], E_DIM, E_DIM);
    P_cwq[i] = ualloc(EE); pack(c_wq + i * EE, P_cwq[i], E_DIM, E_DIM);
    P_cwk[i] = ualloc(EE); pack(c_wk + i * EE, P_cwk[i], E_DIM, E_DIM);
    P_cwv[i] = ualloc(EE); pack(c_wv + i * EE, P_cwv[i], E_DIM, E_DIM);
    P_cwo[i] = ualloc(EE); pack(c_wo + i * EE, P_cwo[i], E_DIM, E_DIM);
    P_cf1[i] = ualloc(EF); pack(c_ffw1 + i * EF, P_cf1[i], E_DIM, FF_DIM);
    P_cf2[i] = ualloc(EF); pack(c_ffw2 + i * EF, P_cf2[i], FF_DIM, E_DIM);
  }
  P_outw = ualloc(packed_elems(E_DIM, V_SIZE)); pack(out_w, P_outw, E_DIM, V_SIZE);

  auto gemm = [&](const float* A, const unsigned short* Bpk, const float* bias,
                  float* C, int M, int N, int K, int relu) {
    dim3 grid((N + 127) / 128, (M + 127) / 128);
    wmma_gemm_kernel<<<grid, dim3(256), 0, stream>>>(A, Bpk, bias, C, M, N, K, relu);
  };
  auto attn = [&](const float* Qa, const float* Ka, const float* Va, float* Oa,
                  int q_len, int k_len, int causal) {
    dim3 grid((q_len + 15) / 16, H_HEADS, NB);
    attn_kernel<<<grid, dim3(32), 0, stream>>>(Qa, Ka, Va, Oa, q_len, k_len, causal);
  };
  auto lnadd = [&](const float* a, const float* b, const float* g, const float* be,
                   float* o2, int rows) {
    ln_add_kernel<<<dim3((rows + 7) / 8), dim3(256), 0, stream>>>(a, b, g, be, o2, rows);
  };

  // ---------------- Encoder ----------------
  gemm(src, P_fcw, fc_b, x0, Me, E_DIM, FC_DIM, 1);
  {
    int tot = Me * E_DIM;
    add_pos_kernel<<<dim3((tot + 255) / 256), dim3(256), 0, stream>>>(x0, enc_pos, Me, S_LEN);
  }
  for (int i = 0; i < L_LAYERS; ++i) {
    gemm(x0, P_ewq[i], nullptr, bq, Me, E_DIM, E_DIM, 0);
    gemm(x0, P_ewk[i], nullptr, bk, Me, E_DIM, E_DIM, 0);
    gemm(x0, P_ewv[i], nullptr, bv, Me, E_DIM, E_DIM, 0);
    attn(bq, bk, bv, ba, S_LEN, S_LEN, 0);
    gemm(ba, P_ewo[i], e_bo + i * E_DIM, bb, Me, E_DIM, E_DIM, 0);
    lnadd(bb, x0, e_ln1g + i * E_DIM, e_ln1b + i * E_DIM, bx1, Me);
    gemm(bx1, P_ef1[i], e_ffb1 + i * FF_DIM, ffh, Me, FF_DIM, E_DIM, 1);
    gemm(ffh, P_ef2[i], e_ffb2 + i * E_DIM, ffo, Me, E_DIM, FF_DIM, 0);
    lnadd(ffo, bx1, e_ln2g + i * E_DIM, e_ln2b + i * E_DIM, x0, Me);
  }

  // ---------------- Decoder ----------------
  {
    int tot = Md * E_DIM;
    embed_kernel<<<dim3((tot + 255) / 256), dim3(256), 0, stream>>>(trg, word_emb, dec_pos, dy, Md);
  }
  for (int i = 0; i < L_LAYERS; ++i) {
    gemm(dy, P_swq[i], nullptr, bq, Md, E_DIM, E_DIM, 0);
    gemm(dy, P_swk[i], nullptr, bk, Md, E_DIM, E_DIM, 0);
    gemm(dy, P_swv[i], nullptr, bv, Md, E_DIM, E_DIM, 0);
    attn(bq, bk, bv, ba, TT, TT, 1);
    gemm(ba, P_swo[i], s_bo + i * E_DIM, bb, Md, E_DIM, E_DIM, 0);
    lnadd(bb, dy, s_lng + i * E_DIM, s_lnb + i * E_DIM, bx1, Md);      // q
    gemm(bx1, P_cwq[i], nullptr, bq, Md, E_DIM, E_DIM, 0);
    gemm(x0,  P_cwk[i], nullptr, bk, Me, E_DIM, E_DIM, 0);
    gemm(x0,  P_cwv[i], nullptr, bv, Me, E_DIM, E_DIM, 0);
    attn(bq, bk, bv, ba, TT, S_LEN, 0);
    gemm(ba, P_cwo[i], c_bo + i * E_DIM, bb, Md, E_DIM, E_DIM, 0);
    lnadd(bb, bx1, c_ln1g + i * E_DIM, c_ln1b + i * E_DIM, ffo, Md);   // x
    gemm(ffo, P_cf1[i], c_ffb1 + i * FF_DIM, ffh, Md, FF_DIM, E_DIM, 1);
    gemm(ffh, P_cf2[i], c_ffb2 + i * E_DIM, bb, Md, E_DIM, FF_DIM, 0);
    lnadd(bb, ffo, c_ln2g + i * E_DIM, c_ln2b + i * E_DIM, dy, Md);
  }

  gemm(dy, P_outw, out_b, out, Md, V_SIZE, E_DIM, 0);
}
